// GaussRenderer_17806934409403
// MI455X (gfx1250) — matrix operand order repaired and checked by reading, not compiled, not run on hardware
//
#include <hip/hip_runtime.h>
#include <hip/hip_bf16.h>
#include <math.h>

typedef __attribute__((ext_vector_type(16))) _Float16 v16h;
typedef __attribute__((ext_vector_type(8)))  _Float16 v8h;
typedef __attribute__((ext_vector_type(8)))  float    v8f;
typedef __attribute__((ext_vector_type(4)))  unsigned int u32x4;
typedef __attribute__((ext_vector_type(8)))  int      i32x8;
typedef __attribute__((ext_vector_type(4)))  int      i32x4;

#define IMG_H 128
#define IMG_W 128
#define TSZ   64
#define NT    4
#define PMAX  2048

// ---------------------------------------------------------------------------
// Kernel 1: per (tile, slot) depth keys: key = overlap ? depth : 1e10 ; pad 1e30
// ---------------------------------------------------------------------------
__global__ void k_init(const float* __restrict__ means, const float* __restrict__ cov,
                       const float* __restrict__ depths,
                       float* __restrict__ keys, int* __restrict__ vals,
                       int N, int sortN) {
    int gid = blockIdx.x * blockDim.x + threadIdx.x;
    int total = NT * sortN;
    if (gid >= total) return;
    int t = gid / sortN, n = gid - t * sortN;
    float key = 1e30f;
    if (n < N) {
        float a = cov[n*4+0], b = cov[n*4+1], c = cov[n*4+2], d = cov[n*4+3];
        float det = a*d - b*c;
        float mid = 0.5f*(a + d);
        float tmp = mid*mid - det; tmp = fmaxf(tmp, 0.1f);
        float s = sqrtf(tmp);
        float lam = fmaxf(mid + s, mid - s);
        float radii = 3.0f * ceilf(sqrtf(lam));
        float mx = means[n*2], my = means[n*2+1];
        float rminx = fminf(fmaxf(mx - radii, 0.f), (float)(IMG_W-1));
        float rmaxx = fminf(fmaxf(mx + radii, 0.f), (float)(IMG_W-1));
        float rminy = fminf(fmaxf(my - radii, 0.f), (float)(IMG_H-1));
        float rmaxy = fminf(fmaxf(my + radii, 0.f), (float)(IMG_H-1));
        int th_ = t >> 1, tw_ = t & 1;            // t = th*TW + tw, TW=2
        float wmin = tw_ * (float)TSZ, wmax = wmin + (TSZ - 1.f);
        float hmin = th_ * (float)TSZ, hmax = hmin + (TSZ - 1.f);
        bool ovx = fminf(rmaxx, wmax) > fmaxf(rminx, wmin);
        bool ovy = fminf(rmaxy, hmax) > fmaxf(rminy, hmin);
        key = (ovx && ovy) ? depths[n] : 1e10f;
    }
    keys[t*sortN + n] = key;
    vals[t*sortN + n] = n;
}

// ---------------------------------------------------------------------------
// Kernel 2: bitonic sort step, lexicographic (key, index) => stable-equivalent
// ---------------------------------------------------------------------------
__global__ void k_bitonic(float* __restrict__ keys, int* __restrict__ vals,
                          int j, int k, int sortN) {
    int gid = blockIdx.x * blockDim.x + threadIdx.x;
    int total = NT * sortN;
    if (gid >= total) return;
    int t = gid / sortN, i = gid - t * sortN;
    int ixj = i ^ j;
    if (ixj <= i) return;
    float* K = keys + (size_t)t * sortN;
    int*   V = vals + (size_t)t * sortN;
    float ka = K[i], kb = K[ixj];
    int   va = V[i], vb = V[ixj];
    bool agtb = (ka > kb) || (ka == kb && va > vb);
    bool asc  = ((i & k) == 0);
    if (asc ? agtb : !agtb) {
        K[i] = kb; K[ixj] = ka;
        V[i] = vb; V[ixj] = va;
    }
}

// ---------------------------------------------------------------------------
// Kernel 3: gather front PMAX per tile -> SoA params (10 arrays of [NT*PMAX])
//   0 mx, 1 my, 2 c00, 3 c01sum, 4 c11, 5 opacity, 6 r, 7 g, 8 b, 9 depth(masked)
// ---------------------------------------------------------------------------
__global__ void k_gather(const float* __restrict__ means, const float* __restrict__ cov,
                         const float* __restrict__ color, const float* __restrict__ opacity,
                         const float* __restrict__ keys, const int* __restrict__ vals,
                         float* __restrict__ gp, int sortN) {
    int gid = blockIdx.x * blockDim.x + threadIdx.x;
    if (gid >= NT * PMAX) return;
    int t = gid >> 11, p = gid & (PMAX - 1);
    float key = keys[(size_t)t * sortN + p];
    int   idx = vals[(size_t)t * sortN + p];
    bool valid = key < 1e9f;
    float mx = 0.f, my = 0.f, c00 = 1.f, c01 = 0.f, c11 = 1.f;
    float op = 0.f, cr = 0.f, cg = 0.f, cb = 0.f, dd = 0.f;
    if (valid) {
        mx = means[idx*2]; my = means[idx*2+1];
        float a = cov[idx*4+0], b = cov[idx*4+1], c = cov[idx*4+2], d = cov[idx*4+3];
        float det = fmaxf(a*d - b*c, 1e-6f);
        float invd = 1.f / det;
        c00 = d * invd;
        c01 = -(b + c) * invd;   // (c01 + c10) of the reference
        c11 = a * invd;
        op = opacity[idx];
        cr = color[idx*3+0]; cg = color[idx*3+1]; cb = color[idx*3+2];
        dd = key;                // masked depth: 0 when invalid
    }
    int o = t * PMAX + p;
    const int S = NT * PMAX;
    gp[0*S+o] = mx;  gp[1*S+o] = my;  gp[2*S+o] = c00; gp[3*S+o] = c01; gp[4*S+o] = c11;
    gp[5*S+o] = op;  gp[6*S+o] = cr;  gp[7*S+o] = cg;  gp[8*S+o] = cb;  gp[9*S+o] = dd;
}

// ---------------------------------------------------------------------------
// Kernel 4: render. 256 thr = 8 waves; wave owns 32 pixels; K chunks of 32.
// Param tiles (10 rows x 32 floats, row stride S) pulled global->LDS by the
// Tensor Data Mover (TDM); serial compositing in f32 VALU; channel
// accumulation via v_wmma_f32_16x16x32_f16.
// Channels (B-matrix N): 0=r 1=g 2=b 3=depth 4=one(acc) 5..15=0
// ---------------------------------------------------------------------------
__global__ __launch_bounds__(256) void k_render(const float* __restrict__ gp,
                                                float* __restrict__ out) {
    __shared__ float sprm[11][32];        // 0..5 geom/op, 6..10 = channels r,g,b,d,1
    __shared__ v8h   wbuf[8][32][4];      // [wave][pixel][k-octet] f16 weights

    const int S = NT * PMAX;
    int tid  = threadIdx.x;
    int wv   = tid >> 5, lane = tid & 31;
    int tile = blockIdx.x >> 4;
    int tp   = ((blockIdx.x & 15) << 8) + tid;        // tile-local pixel 0..4095
    float px = (float)((tile & 1) * TSZ + (tp & 63)); // global x
    float py = (float)((tile >> 1) * TSZ + (tp >> 6));// global y
    int go   = tile * PMAX;

    // constant "ones" channel (acc column of B), staged once
    if (tid < 32) sprm[10][tid] = 1.0f;

    v8f accs[2];
    #pragma unroll
    for (int h = 0; h < 2; ++h)
        #pragma unroll
        for (int e = 0; e < 8; ++e) accs[h][e] = 0.f;

    float T = 1.f;
    int N_ = lane & 15;                   // B column / D channel for this lane
    int kb = (lane < 16) ? 0 : 16;        // B K-base for this lane half
    unsigned int lds_base = (unsigned int)(unsigned long long)(&sprm[0][0]);

    for (int ch = 0; ch < PMAX / 32; ++ch) {
        __syncthreads();                  // prior chunk's sprm/wbuf reads done
        if (wv == 0) {
            // ---- TDM: 2D tile load 32x10 f32, row stride S, into sprm[0..9] ----
            unsigned long long gaddr =
                (unsigned long long)(size_t)(gp + go + ch * 32);
            u32x4 g0;
            g0[0] = 1u;                                        // count=1, load, user
            g0[1] = lds_base;                                  // lds_addr
            g0[2] = (unsigned int)(gaddr & 0xFFFFFFFFu);       // global_addr lo
            g0[3] = (unsigned int)((gaddr >> 32) & 0x01FFFFFFu)
                    | (2u << 30);                              // addr hi | type=2
            i32x8 g1;
            g1[0] = 0x00020000;                 // wg_mask=0, data_size=4B
            g1[1] = (int)(((unsigned)S & 0xFFFFu) << 16);      // tensor_dim0 lo16
            g1[2] = (int)(((unsigned)S >> 16) & 0xFFFFu) | (10 << 16); // dim0 hi | tensor_dim1=10
            g1[3] = 32 << 16;                   // tile_dim0 = 32
            g1[4] = 10;                         // tile_dim1 = 10, tile_dim2 = 0
            g1[5] = S;                          // tensor_dim0_stride lo32
            g1[6] = 0;                          // stride hi16, dim1_stride lo16
            g1[7] = 0;
            i32x4 z4 = {0, 0, 0, 0};
            i32x8 z8 = {0, 0, 0, 0, 0, 0, 0, 0};
            __builtin_amdgcn_tensor_load_to_lds(g0, g1, z4, z4, z8, 0);
            __builtin_amdgcn_s_wait_tensorcnt(0);
        }
        __syncthreads();                  // sprm tile visible to all waves

        // --- serial compositing for this lane's pixel over 32 gaussians ---
        v8h wq[4];
        #pragma unroll
        for (int kk = 0; kk < 32; ++kk) {
            float dx = px - sprm[0][kk];
            float dy = py - sprm[1][kk];
            float pwr = -0.5f * (dx*dx*sprm[2][kk] + dy*dy*sprm[4][kk] + dx*dy*sprm[3][kk]);
            float g  = __expf(pwr);
            float al = fminf(g * sprm[5][kk], 0.99f);   // opacity==0 when invalid
            float w  = al * T;
            T *= (1.f - al);
            wq[kk >> 3][kk & 7] = (_Float16)w;
        }
        wbuf[wv][lane][0] = wq[0];
        wbuf[wv][lane][1] = wq[1];
        wbuf[wv][lane][2] = wq[2];
        wbuf[wv][lane][3] = wq[3];
        __syncthreads();

        // --- A matrices (16x32 f16 layout): lane<16 holds K 0..15, lane>=16 K 16..31 ---
        int q0 = (lane < 16) ? 0 : 2;
        v8h a0lo = wbuf[wv][N_][q0],      a0hi = wbuf[wv][N_][q0 + 1];
        v8h a1lo = wbuf[wv][16 + N_][q0], a1hi = wbuf[wv][16 + N_][q0 + 1];
        v16h A0, A1;
        #pragma unroll
        for (int e = 0; e < 8; ++e) {
            A0[e] = a0lo[e]; A0[8 + e] = a0hi[e];
            A1[e] = a1lo[e]; A1[8 + e] = a1hi[e];
        }
        // --- B matrix (32x16 f16): lane half selects K 0..15 / 16..31 ---
        v16h B;
        #pragma unroll
        for (int j = 0; j < 8; ++j) {
            int k0 = kb + 2 * j;
            float v0 = (N_ < 5) ? sprm[6 + N_][k0]     : 0.f;
            float v1 = (N_ < 5) ? sprm[6 + N_][k0 + 1] : 0.f;
            B[2*j]     = (_Float16)v0;
            B[2*j + 1] = (_Float16)v1;
        }
        accs[0] = __builtin_amdgcn_wmma_f32_16x16x32_f16(false, A0, false, B,
                                                         (short)0, accs[0], false, false);
        accs[1] = __builtin_amdgcn_wmma_f32_16x16x32_f16(false, A1, false, B,
                                                         (short)0, accs[1], false, false);
    }

    // --- write out D: lane = channel N_, rows r -> pixel M ---
    if (N_ < 5) {
        int wbase = ((blockIdx.x & 15) << 8) + wv * 32;
        #pragma unroll
        for (int h = 0; h < 2; ++h) {
            #pragma unroll
            for (int r = 0; r < 8; ++r) {
                int m   = (lane < 16) ? r : (8 + r);
                int tpo = wbase + h * 16 + m;
                int gx  = (tile & 1) * TSZ + (tpo & 63);
                int gy  = (tile >> 1) * TSZ + (tpo >> 6);
                int pix = gy * IMG_W + gx;
                float v = accs[h][r];
                if (N_ < 3)       out[pix * 3 + N_] = v;                     // color
                else if (N_ == 3) out[IMG_H * IMG_W * 3 + pix] = v;          // depth
                else              out[IMG_H * IMG_W * 4 + pix] = v;          // alpha
            }
        }
    }
}

// ---------------------------------------------------------------------------
// Kernel 5: white background: color += (1 - acc)
// ---------------------------------------------------------------------------
__global__ void k_final(float* __restrict__ out) {
    int p = blockIdx.x * blockDim.x + threadIdx.x;
    if (p >= IMG_H * IMG_W) return;
    float wb = 1.f - out[IMG_H * IMG_W * 4 + p];
    out[p*3+0] += wb;
    out[p*3+1] += wb;
    out[p*3+2] += wb;
}

extern "C" void kernel_launch(void* const* d_in, const int* in_sizes, int n_in,
                              void* d_out, int out_size, void* d_ws, size_t ws_size,
                              hipStream_t stream) {
    const float* means   = (const float*)d_in[0];
    const float* cov     = (const float*)d_in[1];
    const float* color   = (const float*)d_in[2];
    const float* opacity = (const float*)d_in[3];
    const float* depths  = (const float*)d_in[4];
    float* out = (float*)d_out;

    int N = in_sizes[0] / 2;
    int sortN = 1; while (sortN < N) sortN <<= 1;

    char* ws   = (char*)d_ws;
    float* keys = (float*)ws;
    int*   vals = (int*)(ws + (size_t)NT * sortN * sizeof(float));
    float* gp   = (float*)(ws + (size_t)2 * NT * sortN * sizeof(float));

    int totalSort = NT * sortN;
    int sb = (totalSort + 255) / 256;

    k_init<<<sb, 256, 0, stream>>>(means, cov, depths, keys, vals, N, sortN);
    for (int k = 2; k <= sortN; k <<= 1)
        for (int j = k >> 1; j > 0; j >>= 1)
            k_bitonic<<<sb, 256, 0, stream>>>(keys, vals, j, k, sortN);
    k_gather<<<(NT * PMAX + 255) / 256, 256, 0, stream>>>(means, cov, color, opacity,
                                                          keys, vals, gp, sortN);
    k_render<<<NT * 16, 256, 0, stream>>>(gp, out);
    k_final<<<(IMG_H * IMG_W + 255) / 256, 256, 0, stream>>>(out);
}
